// TensorVMSplit_32238024524553
// MI455X (gfx1250) — compile-verified, matrix operand order checked by reading, not run on hardware
//
#include <hip/hip_runtime.h>
#include <hip/hip_bf16.h>
#include <stddef.h>

typedef float    f4   __attribute__((ext_vector_type(4)));
typedef _Float16 v4h  __attribute__((ext_vector_type(4)));
typedef _Float16 v8h  __attribute__((ext_vector_type(8)));
typedef _Float16 v16h __attribute__((ext_vector_type(16)));
typedef float    v8f  __attribute__((ext_vector_type(8)));

#define GDIM   300
#define GG     90000
#define NPTS   524288
#define RS     168            // LDS row stride in halves (bank-conflict-free)
#define KPAD   160            // 144 features zero-padded to 5*32

static __device__ __forceinline__ int iclamp(int v, int lo, int hi) {
    return v < lo ? lo : (v > hi ? hi : v);
}

static __device__ __forceinline__ void idxw(float c, int size, int& i0, int& i1, float& f) {
    float x  = (c + 1.0f) * 0.5f * (float)(size - 1);
    float x0 = floorf(x);
    f  = x - x0;
    i0 = iclamp((int)x0, 0, size - 1);
    i1 = iclamp(i0 + 1, 0, size - 1);
}

// ---------- pre-pass: (3,C,G,G) -> (3,G,G,64) channel-last, 64 = 16 density + 48 app
__global__ void __launch_bounds__(128) transpose_planes(const float* __restrict__ dp,
                                                        const float* __restrict__ ap,
                                                        float* __restrict__ outT) {
    __shared__ float tile[64][129];
    const int nb = (GG + 127) / 128;          // 704
    int i      = blockIdx.x / nb;
    int dhbase = (blockIdx.x % nb) * 128;
    int t      = threadIdx.x;
    int dh     = dhbase + t;
    #pragma unroll
    for (int cc = 0; cc < 64; ++cc) {
        float v = 0.0f;
        if (dh < GG)
            v = (cc < 16) ? dp[(size_t)(i * 16 + cc) * GG + dh]
                          : ap[(size_t)(i * 48 + (cc - 16)) * GG + dh];
        tile[cc][t] = v;
    }
    __syncthreads();
    #pragma unroll
    for (int it = 0; it < 64; ++it) {
        int flat = it * 128 + t;
        int cc   = flat & 63;
        int dl_  = flat >> 6;
        int dh2  = dhbase + dl_;
        if (dh2 < GG)
            outT[((size_t)i * GG + dh2) * 64 + cc] = tile[cc][dl_];
    }
}

// ---------- pre-pass: lines (3,C,G) -> (3,G,64)
__global__ void __launch_bounds__(256) transpose_lines(const float* __restrict__ dl,
                                                       const float* __restrict__ al,
                                                       float* __restrict__ outT) {
    int idx = blockIdx.x * 256 + threadIdx.x;
    if (idx >= 3 * GDIM * 64) return;
    int cc   = idx & 63;
    int rest = idx >> 6;
    int d    = rest % GDIM;
    int i    = rest / GDIM;
    float v = (cc < 16) ? dl[(i * 16 + cc) * GDIM + d]
                        : al[(i * 48 + (cc - 16)) * GDIM + d];
    outT[idx] = v;
}

// ---------- load one v16h WMMA fragment (A or B) from an LDS row
static __device__ __forceinline__ v16h ldfrag(const _Float16* rowp, int K0, int h8) {
    v8h lo = *(const v8h*)(rowp + K0 + h8);
    v8h hi = *(const v8h*)(rowp + K0 + 16 + h8);
    v16h r;
    #pragma unroll
    for (int j = 0; j < 8; ++j) { r[j] = lo[j]; r[j + 8] = hi[j]; }
    return r;
}

static __device__ __forceinline__ float hsum4(f4 v) { return v.x + v.y + v.z + v.w; }

// ---------- main kernel: per-point sampling + WMMA feature projection
template<bool CHLAST>
__global__ void __launch_bounds__(256) tvm_main(const float* __restrict__ xyz,
                                                const float* __restrict__ dplane,
                                                const float* __restrict__ dline,
                                                const float* __restrict__ aplane,
                                                const float* __restrict__ aline,
                                                const float* __restrict__ W,
                                                const float* __restrict__ planesT,
                                                const float* __restrict__ linesT,
                                                float* __restrict__ out) {
    extern __shared__ _Float16 smem[];
    _Float16* feats = smem;               // 256 rows x RS halves
    _Float16* Wlds  = smem + 256 * RS;    // 32 rows x RS halves (Wt: row n holds W[n][k])

    const int tid = threadIdx.x;
    const int p   = blockIdx.x * 256 + tid;

    // stage basis_W (27x144 f32) -> padded 32x160 f16 in LDS
    for (int idx = tid; idx < 32 * RS; idx += 256) {
        int n = idx / RS;
        int k = idx - n * RS;
        float v = (n < 27 && k < 144) ? W[n * 144 + k] : 0.0f;
        Wlds[idx] = (_Float16)v;
    }

    float x = xyz[p * 3 + 0];
    float y = xyz[p * 3 + 1];
    float z = xyz[p * 3 + 2];
    float dcs[3] = { z, y, x };
    float hcs[3] = { y, z, z };
    float sigma = 0.0f;

    _Float16* frow = feats + tid * RS;
    #pragma unroll
    for (int k = 144; k < KPAD; ++k) frow[k] = (_Float16)0.0f;   // zero K padding

    #pragma unroll
    for (int i = 0; i < 3; ++i) {
        int d0, d1, h0, h1; float fd, fh;
        idxw(dcs[i], GDIM, d0, d1, fd);
        idxw(hcs[i], GDIM, h0, h1, fh);
        float w00 = (1.0f - fd) * (1.0f - fh);
        float w01 = (1.0f - fd) * fh;
        float w10 = fd * (1.0f - fh);
        float w11 = fd * fh;
        float wl0 = 1.0f - fd, wl1 = fd;

        if (CHLAST) {
            const f4* c00 = (const f4*)(planesT + ((size_t)i * GG + d0 * GDIM + h0) * 64);
            const f4* c01 = (const f4*)(planesT + ((size_t)i * GG + d0 * GDIM + h1) * 64);
            const f4* c10 = (const f4*)(planesT + ((size_t)i * GG + d1 * GDIM + h0) * 64);
            const f4* c11 = (const f4*)(planesT + ((size_t)i * GG + d1 * GDIM + h1) * 64);
            const f4* l0  = (const f4*)(linesT  + ((size_t)i * GDIM + d0) * 64);
            const f4* l1  = (const f4*)(linesT  + ((size_t)i * GDIM + d1) * 64);
            #pragma unroll
            for (int c4 = 0; c4 < 16; ++c4) {
                f4 pv = w00 * c00[c4] + w01 * c01[c4] + w10 * c10[c4] + w11 * c11[c4];
                f4 lv = wl0 * l0[c4] + wl1 * l1[c4];
                f4 fv = pv * lv;
                if (c4 < 4) {
                    sigma += hsum4(fv);
                } else {
                    int k = i * 48 + (c4 - 4) * 4;
                    v4h hv;
                    hv[0] = (_Float16)fv.x; hv[1] = (_Float16)fv.y;
                    hv[2] = (_Float16)fv.z; hv[3] = (_Float16)fv.w;
                    *(v4h*)(frow + k) = hv;
                }
            }
        } else {
            for (int c = 0; c < 16; ++c) {
                const float* P = dplane + (size_t)(i * 16 + c) * GG;
                const float* L = dline + (i * 16 + c) * GDIM;
                float pv = w00 * P[d0 * GDIM + h0] + w01 * P[d0 * GDIM + h1]
                         + w10 * P[d1 * GDIM + h0] + w11 * P[d1 * GDIM + h1];
                sigma += pv * (wl0 * L[d0] + wl1 * L[d1]);
            }
            for (int c = 0; c < 48; ++c) {
                const float* P = aplane + (size_t)(i * 48 + c) * GG;
                const float* L = aline + (i * 48 + c) * GDIM;
                float pv = w00 * P[d0 * GDIM + h0] + w01 * P[d0 * GDIM + h1]
                         + w10 * P[d1 * GDIM + h0] + w11 * P[d1 * GDIM + h1];
                float fv = pv * (wl0 * L[d0] + wl1 * L[d1]);
                frow[i * 48 + c] = (_Float16)fv;
            }
        }
    }
    out[p] = sigma;
    __syncthreads();

    // ---- WMMA phase: (256 x 160) f16 feats  @  (160 x 32) f16 Wt -> (256 x 32) f32
    const int lane = tid & 31;
    const int wid  = tid >> 5;
    const int h8   = (lane >> 4) << 3;     // 0 or 8 (K sub-block per half-wave)
    const int r16  = lane & 15;

    const _Float16* arow0 = feats + (wid * 32 + r16) * RS;
    const _Float16* arow1 = arow0 + 16 * RS;
    const _Float16* brow0 = Wlds + r16 * RS;
    const _Float16* brow1 = Wlds + (r16 + 16) * RS;

    v8f acc00 = {0,0,0,0,0,0,0,0};
    v8f acc01 = {0,0,0,0,0,0,0,0};
    v8f acc10 = {0,0,0,0,0,0,0,0};
    v8f acc11 = {0,0,0,0,0,0,0,0};

    #pragma unroll
    for (int ks = 0; ks < 5; ++ks) {
        int K0 = ks * 32;
        v16h a0 = ldfrag(arow0, K0, h8);
        v16h a1 = ldfrag(arow1, K0, h8);
        v16h b0 = ldfrag(brow0, K0, h8);
        v16h b1 = ldfrag(brow1, K0, h8);
        acc00 = __builtin_amdgcn_wmma_f32_16x16x32_f16(false, a0, false, b0, (short)0, acc00, false, false);
        acc01 = __builtin_amdgcn_wmma_f32_16x16x32_f16(false, a0, false, b1, (short)0, acc01, false, false);
        acc10 = __builtin_amdgcn_wmma_f32_16x16x32_f16(false, a1, false, b0, (short)0, acc10, false, false);
        acc11 = __builtin_amdgcn_wmma_f32_16x16x32_f16(false, a1, false, b1, (short)0, acc11, false, false);
    }

    // C/D layout: VGPR r, lanes 0-15 -> M=r, N=lane; lanes 16-31 -> M=r+8, N=lane-16
    float* appout = out + NPTS;
    int gmBase = blockIdx.x * 256 + wid * 32 + ((lane >> 4) << 3);
    int n0 = r16;
    int n1 = 16 + r16;
    #pragma unroll
    for (int r = 0; r < 8; ++r) {
        int m0 = gmBase + r;        // M-tile 0
        int m1 = m0 + 16;           // M-tile 1
        appout[(size_t)m0 * 27 + n0] = acc00[r];
        appout[(size_t)m1 * 27 + n0] = acc10[r];
        if (n1 < 27) {
            appout[(size_t)m0 * 27 + n1] = acc01[r];
            appout[(size_t)m1 * 27 + n1] = acc11[r];
        }
    }
}

extern "C" void kernel_launch(void* const* d_in, const int* in_sizes, int n_in,
                              void* d_out, int out_size, void* d_ws, size_t ws_size,
                              hipStream_t stream) {
    const float* xyz = (const float*)d_in[0];
    const float* dp  = (const float*)d_in[1];
    const float* dl  = (const float*)d_in[2];
    const float* ap  = (const float*)d_in[3];
    const float* al  = (const float*)d_in[4];
    const float* W   = (const float*)d_in[5];
    float* out = (float*)d_out;

    const size_t needP = (size_t)3 * GG * 64 * sizeof(float);     // ~69.1 MB
    const size_t needL = (size_t)3 * GDIM * 64 * sizeof(float);
    const size_t shmem = (size_t)(256 + 32) * RS * 2;             // ~94.5 KB f16 LDS

    const int nblk = NPTS / 256;

    if (ws_size >= needP + needL) {
        float* planesT = (float*)d_ws;
        float* linesT  = planesT + (size_t)3 * GG * 64;
        const int nb = (GG + 127) / 128;
        transpose_planes<<<3 * nb, 128, 0, stream>>>(dp, ap, planesT);
        transpose_lines<<<(3 * GDIM * 64 + 255) / 256, 256, 0, stream>>>(dl, al, linesT);
        tvm_main<true><<<nblk, 256, shmem, stream>>>(xyz, dp, dl, ap, al, W, planesT, linesT, out);
    } else {
        tvm_main<false><<<nblk, 256, shmem, stream>>>(xyz, dp, dl, ap, al, W, nullptr, nullptr, out);
    }
}
